// GNN_395136991532
// MI455X (gfx1250) — matrix operand order repaired and checked by reading, not compiled, not run on hardware
//
#include <hip/hip_runtime.h>

// ---------------- problem constants ----------------
constexpr int kN = 50000;   // nodes
constexpr int kE = 600000;  // edges
constexpr int kG = 256;     // graphs
constexpr int kD = 128;     // hidden dim
constexpr int kL = 5;       // layers
constexpr int kC = 10;      // classes

typedef __attribute__((ext_vector_type(16))) __bf16 v16bf;
typedef __attribute__((ext_vector_type(8)))  __bf16 v8bf;
typedef __attribute__((ext_vector_type(8)))  float  v8f;

// ---------------- elementwise helpers ----------------
__global__ __launch_bounds__(256) void k_init_h(const float* atom_emb, const int* x,
                                                float* h, long n) {
    long gid = (long)blockIdx.x * blockDim.x + threadIdx.x;
    if (gid >= n) return;
    int i = (int)(gid / kD), d = (int)(gid % kD);
    h[gid] = atom_emb[(size_t)x[i] * kD + d];
}

__global__ __launch_bounds__(256) void k_init_vn(const float* vn_emb, float* vn, long n) {
    long gid = (long)blockIdx.x * blockDim.x + threadIdx.x;
    if (gid >= n) return;
    vn[gid] = vn_emb[gid % kD];
}

__global__ __launch_bounds__(256) void k_fill0(float* p, long n) {
    long gid = (long)blockIdx.x * blockDim.x + threadIdx.x;
    if (gid < n) p[gid] = 0.0f;
}

__global__ __launch_bounds__(256) void k_copy(float* dstp, const float* srcp, long n) {
    long gid = (long)blockIdx.x * blockDim.x + threadIdx.x;
    if (gid < n) dstp[gid] = srcp[gid];
}

__global__ __launch_bounds__(256) void k_cvt_bf16(const float* in, __bf16* out, long n) {
    long gid = (long)blockIdx.x * blockDim.x + threadIdx.x;
    if (gid < n) out[gid] = (__bf16)in[gid];
}

// Pack f32 weight matrix B[K x Nn] (row-major) into bf16 WMMA B-fragments.
// Fragment f = kb*(Nn/16) + tn holds 512 halves: [lane 0..31][half 0..15], with
//   hi=lane>>4, lo=lane&15, element <- B[(kb*32 + (h<8 ? hi*8+h : 16+hi*8+h-8))*Nn + tn*16 + lo].
// GEMM loads each lane's 16 halves as one contiguous 32-byte vector.
__global__ __launch_bounds__(256) void k_pack_b(const float* B, __bf16* Bp, int K, int Nn) {
    long gid = (long)blockIdx.x * blockDim.x + threadIdx.x;
    long total = (long)K * Nn;
    if (gid >= total) return;
    int h    = (int)(gid & 15);
    int lane = (int)((gid >> 4) & 31);
    int f    = (int)(gid >> 9);
    int tilesN = Nn >> 4;
    int tn = f % tilesN, kb = f / tilesN;
    int hi = lane >> 4, lo = lane & 15;
    int k = kb * 32 + (h < 8 ? hi * 8 + h : 16 + hi * 8 + (h - 8));
    Bp[gid] = (__bf16)B[(size_t)k * Nn + tn * 16 + lo];
}

// h_in = h + vn[batch];  z = (1+eps[l]) * h_in   (agg accumulated into z by atomics)
__global__ __launch_bounds__(256) void k_hin_z(const float* h, const float* vn,
                                               const int* batch, const float* eps, int l,
                                               float* hin, float* z, long n) {
    long gid = (long)blockIdx.x * blockDim.x + threadIdx.x;
    if (gid >= n) return;
    int i = (int)(gid / kD), d = (int)(gid % kD);
    float v = h[gid] + vn[(size_t)batch[i] * kD + d];
    hin[gid] = v;
    z[gid] = (1.0f + eps[l]) * v;
}

// one wave32 per edge: z[dst] += hin[src]  (128 floats = 32 lanes x float4)
__global__ __launch_bounds__(256) void k_edge_scatter(const float* hin, const int* srcI,
                                                      const int* dstI, float* z, int E) {
    int gid = blockIdx.x * blockDim.x + threadIdx.x;
    int e = gid >> 5, lane = gid & 31;
    if (e >= E) return;
    int s = srcI[e], d = dstI[e];
    const float4 v = *(const float4*)(hin + (size_t)s * kD + lane * 4);
    float* zp = z + (size_t)d * kD + lane * 4;
    atomicAdd(zp + 0, v.x); atomicAdd(zp + 1, v.y);
    atomicAdd(zp + 2, v.z); atomicAdd(zp + 3, v.w);
}

// one wave32 per node: out[batch[i]] += rows[i]
__global__ __launch_bounds__(256) void k_node_scatter(const float* rows, const int* batch,
                                                      float* out, int Nn) {
    int gid = blockIdx.x * blockDim.x + threadIdx.x;
    int i = gid >> 5, lane = gid & 31;
    if (i >= Nn) return;
    int g = batch[i];
    const float4 v = *(const float4*)(rows + (size_t)i * kD + lane * 4);
    float* op = out + (size_t)g * kD + lane * 4;
    atomicAdd(op + 0, v.x); atomicAdd(op + 1, v.y);
    atomicAdd(op + 2, v.z); atomicAdd(op + 3, v.w);
}

__global__ __launch_bounds__(256) void k_count(const int* batch, float* cnt, int Nn) {
    int gid = blockIdx.x * blockDim.x + threadIdx.x;
    if (gid < Nn) atomicAdd(&cnt[batch[gid]], 1.0f);
}

// out[g,c] = (hgsum[g]/max(cnt,1)) . predW[:,c] + predb[c]
__global__ __launch_bounds__(256) void k_pred(const float* hgsum, const float* cnt,
                                              const float* predW, const float* predb,
                                              float* out, int Gn) {
    int gid = blockIdx.x * blockDim.x + threadIdx.x;
    if (gid >= Gn * kC) return;
    int g = gid / kC, c = gid % kC;
    float inv = 1.0f / fmaxf(cnt[g], 1.0f);
    float acc = predb[c];
    for (int d = 0; d < kD; ++d)
        acc += hgsum[(size_t)g * kD + d] * inv * predW[(size_t)d * kC + c];
    out[gid] = acc;
}

// ---------------- WMMA bf16 GEMM + fused bias/BN/ReLU ----------------
// C = BN(A(MxK) @ B(KxNN) + bias) [ReLU optional], B pre-packed by k_pack_b.
// K and NN are compile-time: all fragment addresses are base + immediate offset.
// One wave32 computes a 16x64 output strip (4 N-tiles): each A fragment is
// loaded once (2x b128) and reused against 4 packed-B fragments (contiguous
// 32B vector loads) -> 4 v_wmma per K-step, K-loop fully unrolled.
// A fragment (16-bit A 16x32, ISA 7.12.2): hi=lane>>4, m=lane&15,
//   halves 0..7  <- K = kt + hi*8 + 0..7 ; halves 8..15 <- K = kt+16+hi*8+0..7
// C fragment (f32 16x16): vgpr r -> row = r + 8*hi, col = lane&15
template <int K, int NN>
__global__ __launch_bounds__(256) void k_wmma_gemm_bn(
    const __bf16* __restrict__ A, const __bf16* __restrict__ Bp,
    const float* __restrict__ bias, const float* __restrict__ gamma,
    const float* __restrict__ beta, const float* __restrict__ mean,
    const float* __restrict__ var, float* outF, __bf16* outB,
    int M, int relu) {
    constexpr int tilesN  = NN >> 4;   // 16-wide tiles
    constexpr int groupsN = NN >> 6;   // strips of 4 tiles
    int wave = (blockIdx.x * blockDim.x + threadIdx.x) >> 5;
    int lane = threadIdx.x & 31;
    int total = (M >> 4) * groupsN;
    if (wave >= total) return;             // wave-uniform guard (EXEC stays all-1)
    int tm = wave / groupsN, tng = wave % groupsN;
    int hi = lane >> 4, lo = lane & 15;

    v8f c[4] = {{}, {}, {}, {}};
    // hoisted bases: inner loop is load(imm-offset) -> wmma only
    const __bf16* abase = A + (size_t)(tm * 16 + lo) * K + hi * 8;
    const __bf16* bbase = Bp + ((size_t)tng * 4) * 512 + lane * 16;
#pragma unroll
    for (int kt = 0; kt < K; kt += 32) {
        v16bf a;
        {
            v8bf a0 = *(const v8bf*)(abase + kt);
            v8bf a1 = *(const v8bf*)(abase + kt + 16);
#pragma unroll
            for (int i = 0; i < 8; ++i) { a[i] = a0[i]; a[8 + i] = a1[i]; }
        }
        const int kb = kt >> 5;
#pragma unroll
        for (int j = 0; j < 4; ++j) {
            v16bf b = *(const v16bf*)(bbase + (kb * tilesN + j) * 512);
            c[j] = __builtin_amdgcn_wmma_f32_16x16x32_bf16(
                       false, a, false, b, (short)0, c[j], false, false);
        }
    }

    // fused epilogue: bias + BatchNorm(eval) + optional ReLU (col-indexed params)
#pragma unroll
    for (int j = 0; j < 4; ++j) {
        int col = (tng * 4 + j) * 16 + lo;
        float bi = bias[col];
        float ga = gamma[col], be = beta[col], mu = mean[col];
        float rs = rsqrtf(var[col] + 1e-5f);
#pragma unroll
        for (int r = 0; r < 8; ++r) {
            int row = tm * 16 + r + 8 * hi;
            float v = (c[j][r] + bi - mu) * ga * rs + be;
            if (relu) v = fmaxf(v, 0.0f);
            if (outF) outF[(size_t)row * NN + col] = v;
            if (outB) outB[(size_t)row * NN + col] = (__bf16)v;
        }
    }
}

// ---------------- launcher ----------------
static inline int blocks_for(long n) { return (int)((n + 255) / 256); }
static inline int gemm_blocks(int M, int Nn) {
    long waves = (long)(M / 16) * (Nn / 64);
    return (int)((waves * 32 + 255) / 256);
}

extern "C" void kernel_launch(void* const* d_in, const int* in_sizes, int n_in,
                              void* d_out, int out_size, void* d_ws, size_t ws_size,
                              hipStream_t stream) {
    const float* atom_emb = (const float*)d_in[0];
    const float* vn_emb   = (const float*)d_in[1];
    const float* eps      = (const float*)d_in[2];
    const float* W1  = (const float*)d_in[3];
    const float* b1  = (const float*)d_in[4];
    const float* g1  = (const float*)d_in[5];
    const float* be1 = (const float*)d_in[6];
    const float* m1  = (const float*)d_in[7];
    const float* v1  = (const float*)d_in[8];
    const float* W2  = (const float*)d_in[9];
    const float* b2  = (const float*)d_in[10];
    const float* gc  = (const float*)d_in[11];
    const float* bc  = (const float*)d_in[12];
    const float* mc  = (const float*)d_in[13];
    const float* vc  = (const float*)d_in[14];
    const float* vW1 = (const float*)d_in[15];
    const float* vb1 = (const float*)d_in[16];
    const float* vg1 = (const float*)d_in[17];
    const float* vbe1= (const float*)d_in[18];
    const float* vm1 = (const float*)d_in[19];
    const float* vv1 = (const float*)d_in[20];
    const float* vW2 = (const float*)d_in[21];
    const float* vb2 = (const float*)d_in[22];
    const float* vg2 = (const float*)d_in[23];
    const float* vbe2= (const float*)d_in[24];
    const float* vm2 = (const float*)d_in[25];
    const float* vv2 = (const float*)d_in[26];
    const float* predW = (const float*)d_in[27];
    const float* predb = (const float*)d_in[28];
    const int* x     = (const int*)d_in[29];
    const int* ei    = (const int*)d_in[30];
    const int* batch = (const int*)d_in[31];
    const int* esrc = ei;
    const int* edst = ei + kE;
    float* out = (float*)d_out;

    // workspace carve (256B aligned)
    char* ws = (char*)d_ws;
    size_t off = 0;
    auto carve = [&](size_t bytes) -> char* {
        char* p = ws + off;
        off += (bytes + 255) & ~(size_t)255;
        return p;
    };
    float*  h        = (float*)carve(sizeof(float) * (size_t)kN * kD);
    float*  hin      = (float*)carve(sizeof(float) * (size_t)kN * kD);
    float*  z        = (float*)carve(sizeof(float) * (size_t)kN * kD);
    __bf16* zbf      = (__bf16*)carve(2 * (size_t)kN * kD);
    __bf16* t1bf     = (__bf16*)carve(2 * (size_t)kN * 2 * kD);
    float*  vn       = (float*)carve(sizeof(float) * (size_t)kG * kD);
    float*  pooled   = (float*)carve(sizeof(float) * (size_t)kG * kD);
    __bf16* pooledbf = (__bf16*)carve(2 * (size_t)kG * kD);
    __bf16* tbf      = (__bf16*)carve(2 * (size_t)kG * 2 * kD);
    __bf16* wbfA     = (__bf16*)carve(2 * (size_t)kD * 2 * kD);   // packed B (K=128,N=256)
    __bf16* wbfB     = (__bf16*)carve(2 * (size_t)2 * kD * kD);   // packed B (K=256,N=128)
    float*  hgsum    = (float*)carve(sizeof(float) * (size_t)kG * kD);
    float*  cnt      = (float*)carve(sizeof(float) * (size_t)kG);

    const long nND  = (long)kN * kD;
    const long nGD  = (long)kG * kD;
    const long nW1  = (long)kD * 2 * kD;
    const long nW2  = (long)2 * kD * kD;

    // h0 = atom_emb[x]; vn0 broadcast
    k_init_h<<<blocks_for(nND), 256, 0, stream>>>(atom_emb, x, h, nND);
    k_init_vn<<<blocks_for(nGD), 256, 0, stream>>>(vn_emb, vn, nGD);

    for (int l = 0; l < kL; ++l) {
        // h_in and z-init
        k_hin_z<<<blocks_for(nND), 256, 0, stream>>>(h, vn, batch, eps, l, hin, z, nND);
        // GIN aggregation: z[dst] += hin[src]   (HBM-bound scatter, ~614 MB/layer)
        k_edge_scatter<<<blocks_for((long)kE * 32), 256, 0, stream>>>(hin, esrc, edst, z, kE);
        // MLP1: t1 = relu(bn1(z @ W1 + b1))   [WMMA bf16, M=50000 K=128 N=256]
        k_cvt_bf16<<<blocks_for(nND), 256, 0, stream>>>(z, zbf, nND);
        k_pack_b<<<blocks_for(nW1), 256, 0, stream>>>(W1 + (size_t)l * nW1, wbfA, kD, 2 * kD);
        k_wmma_gemm_bn<kD, 2 * kD><<<gemm_blocks(kN, 2 * kD), 256, 0, stream>>>(
            zbf, wbfA, b1 + l * 2 * kD, g1 + l * 2 * kD, be1 + l * 2 * kD,
            m1 + l * 2 * kD, v1 + l * 2 * kD, nullptr, t1bf, kN, 1);
        // MLP2: h = bn(t1 @ W2 + b2) (+relu unless last)  [M=50000 K=256 N=128]
        k_pack_b<<<blocks_for(nW2), 256, 0, stream>>>(W2 + (size_t)l * nW2, wbfB, 2 * kD, kD);
        k_wmma_gemm_bn<2 * kD, kD><<<gemm_blocks(kN, kD), 256, 0, stream>>>(
            t1bf, wbfB, b2 + l * kD, gc + l * kD, bc + l * kD,
            mc + l * kD, vc + l * kD, h, nullptr, kN, (l < kL - 1) ? 1 : 0);
        // virtual-node update (not on last layer)
        if (l < kL - 1) {
            k_copy<<<blocks_for(nGD), 256, 0, stream>>>(pooled, vn, nGD);
            k_node_scatter<<<blocks_for((long)kN * 32), 256, 0, stream>>>(hin, batch, pooled, kN);
            k_cvt_bf16<<<blocks_for(nGD), 256, 0, stream>>>(pooled, pooledbf, nGD);
            k_pack_b<<<blocks_for(nW1), 256, 0, stream>>>(vW1 + (size_t)l * nW1, wbfA, kD, 2 * kD);
            k_wmma_gemm_bn<kD, 2 * kD><<<gemm_blocks(kG, 2 * kD), 256, 0, stream>>>(
                pooledbf, wbfA, vb1 + l * 2 * kD, vg1 + l * 2 * kD, vbe1 + l * 2 * kD,
                vm1 + l * 2 * kD, vv1 + l * 2 * kD, nullptr, tbf, kG, 1);
            k_pack_b<<<blocks_for(nW2), 256, 0, stream>>>(vW2 + (size_t)l * nW2, wbfB, 2 * kD, kD);
            k_wmma_gemm_bn<2 * kD, kD><<<gemm_blocks(kG, kD), 256, 0, stream>>>(
                tbf, wbfB, vb2 + l * kD, vg2 + l * kD, vbe2 + l * kD,
                vm2 + l * kD, vv2 + l * kD, vn, nullptr, kG, 1);
        }
    }

    // global mean pool + prediction head
    k_fill0<<<blocks_for(nGD), 256, 0, stream>>>(hgsum, nGD);
    k_fill0<<<blocks_for((long)kG), 256, 0, stream>>>(cnt, kG);
    k_node_scatter<<<blocks_for((long)kN * 32), 256, 0, stream>>>(h, batch, hgsum, kN);
    k_count<<<blocks_for((long)kN), 256, 0, stream>>>(batch, cnt, kN);
    k_pred<<<blocks_for((long)kG * kC), 256, 0, stream>>>(hgsum, cnt, predW, predb, out, kG);
}
